// SearchTransfer_83416854823298
// MI455X (gfx1250) — matrix operand order repaired and checked by reading, not compiled, not run on hardware
//
#include <hip/hip_runtime.h>
#include <math.h>

// ---------------------------------------------------------------------------
// SearchTransfer for MI455X (gfx1250, wave32, WMMA + TDM).
//   B=64, C=2048, H=W=7, L=49 (pad to 64), CU=C*9=18432.
// Pipeline:
//   1) sumsq + norms (channel sum-of-squares -> 3x3 window -> sqrt, eps clamp)
//   2) corr_argmax: per-batch P = ref^T tgt over C=2048 via f32 WMMA 16x16x4,
//      stages 64-channel slabs via tensor_load_to_lds (TDM OOB zero-fills
//      pixels >= 49), then R = 3x3 shift-sum of P, normalized; max/argmax.
//   3) gather rr (bf16, [b][n64][CU]) from argmax indices
//   4) GEMM1 (bf16 WMMA 16x16x32, TDM double-buffered 64x128 LDS tiles,
//      4 WMMA per barrier): flod_w @ rr + bias -> bf16 into cat (fused concat)
//   5) GEMM2 (same structure): conv_w @ cat + bias, *mask + target -> f32 out
// Workspace: ~277 MB carved from d_ws.
// ---------------------------------------------------------------------------

#define BATCH 64
#define CHN   2048
#define LL    49
#define LP    64
#define CUK   18432
#define CATK  4096
#define KSTG  128            // K elements staged per TDM tile in the GEMMs
#define EPSV  1e-12f

typedef __attribute__((ext_vector_type(16))) __bf16 v16bf;
typedef __attribute__((ext_vector_type(8)))  float  v8f;
typedef __attribute__((ext_vector_type(2)))  float  v2f;
typedef __attribute__((ext_vector_type(4)))  unsigned int v4u;
typedef __attribute__((ext_vector_type(4)))  unsigned int u32x4;
typedef __attribute__((ext_vector_type(8)))  int i32x8;
typedef __attribute__((ext_vector_type(4)))  int i32x4;

// ---- Tensor Data Mover: 2D tile Global -> LDS ------------------------------
// D# packing per CDNA5 ISA (groups 2/3 unused/zero).
// data_size_log2: 0=1B, 1=2B, 2=4B. Dims/strides in data_size units.
__device__ __forceinline__ void tdm_load_2d(
    unsigned lds_off, const void* gptr,
    unsigned tensor_d0, unsigned tensor_d1,
    unsigned tile_d0, unsigned tile_d1,
    unsigned long long stride0, int data_size_log2) {
  unsigned long long ga = (unsigned long long)gptr;
  u32x4 g0;
  g0.x = 1u;                                        // count=1, user mode
  g0.y = lds_off;                                   // lds_addr (bytes)
  g0.z = (unsigned)(ga & 0xffffffffu);              // global_addr[31:0]
  g0.w = (unsigned)((ga >> 32) & 0x1ffffffu)        // global_addr[56:32]
         | (2u << 30);                              // type = 2 ("image")
  i32x8 g1;
  g1[0] = (data_size_log2 << 16);                   // wg_mask=0, data_size
  g1[1] = (int)((tensor_d0 & 0xffffu) << 16);       // atomic_addr=0 | td0 lo
  g1[2] = (int)((tensor_d0 >> 16) | ((tensor_d1 & 0xffffu) << 16));
  g1[3] = (int)((tensor_d1 >> 16) | (tile_d0 << 16));
  g1[4] = (int)(tile_d1 & 0xffffu);                 // tile_dim1 | tile_dim2=0
  g1[5] = (int)(stride0 & 0xffffffffull);           // tensor_dim0_stride lo
  g1[6] = (int)((stride0 >> 32) & 0xffffull);       // stride hi | dim1_stride=0
  g1[7] = 0;
  i32x4 z4 = {0, 0, 0, 0};
#if __clang_major__ >= 23
  i32x8 z8 = {0, 0, 0, 0, 0, 0, 0, 0};
  __builtin_amdgcn_tensor_load_to_lds(g0, g1, z4, z4, z8, 0);
#else
  __builtin_amdgcn_tensor_load_to_lds(g0, g1, z4, z4, 0);
#endif
}

// ---- fragment loaders from LDS (16-bit WMMA 16x16x32, contiguous k = 16h + s in A and B) -
// A 16x32: lanes 0-15 hold K {0..15};        lanes 16-31 hold K {16..31}
// B 32x16: lanes 0-15 hold K {0..15};        lanes 16-31 hold K {16..31}
__device__ __forceinline__ v16bf frag_from_lds_A(const __bf16* p) {
  union { v16bf v; v4u u[2]; } f;
  f.u[0] = *(const v4u*)(p);
  f.u[1] = *(const v4u*)(p + 8);
  return f.v;
}
__device__ __forceinline__ v16bf frag_from_lds_B(const __bf16* p) {
  union { v16bf v; v4u u[2]; } f;
  f.u[0] = *(const v4u*)(p);
  f.u[1] = *(const v4u*)(p + 8);
  return f.v;
}

// ---- 1a: per-pixel channel sum of squares (one wave per (b, tensor, pixel))
__global__ void st_sumsq(const float* __restrict__ refp,
                         const float* __restrict__ tgtp,
                         float* __restrict__ s2) {
  int wid  = (blockIdx.x * blockDim.x + threadIdx.x) >> 5;
  int lane = threadIdx.x & 31;
  if (wid >= BATCH * 2 * LL) return;
  int b = wid / (2 * LL);
  int rem = wid % (2 * LL);
  int t = rem / LL, l = rem % LL;
  const float* src = t ? tgtp : refp;
  float ss = 0.f;
  for (int c = lane; c < CHN; c += 32) {
    float v = src[((size_t)(b * CHN + c)) * LL + l];
    ss += v * v;
  }
  for (int off = 16; off; off >>= 1) ss += __shfl_xor(ss, off, 32);
  if (lane == 0) s2[wid] = ss;
}

// ---- 1b: 3x3 window sum -> norm denominator max(sqrt, eps) -----------------
__global__ void st_norms(const float* __restrict__ s2, float* __restrict__ nrm) {
  int id = blockIdx.x * blockDim.x + threadIdx.x;
  if (id >= BATCH * 2 * LL) return;
  int l = id % LL, h = l / 7, w = l % 7;
  int base = id - l;
  float acc = 0.f;
  for (int dh = -1; dh <= 1; ++dh)
    for (int dw = -1; dw <= 1; ++dw) {
      int hh = h + dh, ww = w + dw;
      if (hh >= 0 && hh < 7 && ww >= 0 && ww < 7) acc += s2[base + hh * 7 + ww];
    }
  nrm[id] = fmaxf(sqrtf(acc), EPSV);
}

// ---- 2: correlation (f32 WMMA, TDM-staged) + 3x3 shift-sum + argmax --------
// One block per batch, 16 waves, each wave owns one 16x16 tile of P (64x64).
// Dynamic LDS (48KB): refs [64][64] f32 @0, tgts @16384, P [64][64] @32768.
__global__ __launch_bounds__(512) void st_corr_argmax(
    const float* __restrict__ refp, const float* __restrict__ tgtp,
    const float* __restrict__ nrm, float* __restrict__ maxv,
    int* __restrict__ maxi) {
  extern __shared__ char smem[];
  float* refs = (float*)smem;
  float* tgts = (float*)(smem + 16384);
  float* P    = (float*)(smem + 32768);
  int b = blockIdx.x;
  int tid = threadIdx.x, lane = tid & 31, w = tid >> 5;
  int mt = w >> 2, nt = w & 3;
  int lm = lane & 15;
  int ko = (lane >> 4) << 1;  // f32 WMMA A 16x4: lanes 0-15 K={0,1}, 16-31 K={2,3}
  v8f acc = {};
  for (int kc = 0; kc < CHN; kc += 64) {
    if (w == 0) {
      // 64 channel-rows x 64 pixel-cols; tensor_dim0=49 -> HW zero-fills n>=49.
      tdm_load_2d(0,     refp + ((size_t)(b * CHN + kc)) * LL,
                  LL, 64, LP, 64, LL, 2);
      tdm_load_2d(16384, tgtp + ((size_t)(b * CHN + kc)) * LL,
                  LL, 64, LP, 64, LL, 2);
      __builtin_amdgcn_s_wait_tensorcnt(0);
    }
    __syncthreads();
#pragma unroll 4
    for (int k4 = 0; k4 < 64; k4 += 4) {
      v2f a, bb;
      a.x  = refs[(k4 + ko)     * LP + mt * 16 + lm];
      a.y  = refs[(k4 + ko + 1) * LP + mt * 16 + lm];
      bb.x = tgts[(k4 + ko)     * LP + nt * 16 + lm];
      bb.y = tgts[(k4 + ko + 1) * LP + nt * 16 + lm];
      acc = __builtin_amdgcn_wmma_f32_16x16x4_f32(
          false, a, false, bb, (short)0, acc, false, false);
    }
    __syncthreads();
  }
  {
    int n = nt * 16 + lm;
    int mbase = mt * 16 + ((lane >> 4) << 3);
    for (int r = 0; r < 8; ++r) P[(mbase + r) * LP + n] = acc[r];
  }
  __syncthreads();
  if (tid < LL) {
    int lt = tid, ht = lt / 7, wt = lt % 7;
    float dt = nrm[(b * 2 + 1) * LL + lt];
    float best = -1e30f; int bi = 0;
    for (int lr = 0; lr < LL; ++lr) {
      int hr = lr / 7, wr = lr % 7;
      float raw = 0.f;
      for (int dh = -1; dh <= 1; ++dh)
        for (int dw = -1; dw <= 1; ++dw) {
          int h1 = hr + dh, w1 = wr + dw, h2 = ht + dh, w2 = wt + dw;
          if (h1 >= 0 && h1 < 7 && w1 >= 0 && w1 < 7 &&
              h2 >= 0 && h2 < 7 && w2 >= 0 && w2 < 7)
            raw += P[(h1 * 7 + w1) * LP + (h2 * 7 + w2)];
        }
      float dr = nrm[(b * 2 + 0) * LL + lr];
      float rv = raw / (dr * dt);
      if (rv > best) { best = rv; bi = lr; }
    }
    maxv[b * LL + lt] = best;
    maxi[b * LL + lt] = bi;
  }
}

// ---- 3a: generic f32 -> bf16 conversion ------------------------------------
__global__ void st_cvt_bf16(const float* __restrict__ in,
                            __bf16* __restrict__ outp, long long n) {
  long long id = (long long)blockIdx.x * blockDim.x + threadIdx.x;
  if (id < n) outp[id] = (__bf16)in[id];
}

// ---- 3b: gather rr[b][n64][CU] bf16 from argmax indices --------------------
__global__ void st_gather_rr(const float* __restrict__ refp,
                             const int* __restrict__ maxi,
                             __bf16* __restrict__ rr) {
  long long id = (long long)blockIdx.x * blockDim.x + threadIdx.x;
  if (id >= (long long)BATCH * LP * CUK) return;
  int k = (int)(id % CUK);
  int n = (int)((id / CUK) % LP);
  int b = (int)(id / ((long long)CUK * LP));
  float val = 0.f;
  if (n < LL) {
    int idx = maxi[b * LL + n];
    int hi = idx / 7, wi = idx % 7;
    int c = k / 9, r9 = k % 9, di = r9 / 3, dj = r9 % 3;
    int hh = hi + di - 1, ww = wi + dj - 1;
    if (hh >= 0 && hh < 7 && ww >= 0 && ww < 7)
      val = refp[((size_t)(b * CHN + c)) * LL + hh * 7 + ww];
  }
  rr[id] = (__bf16)val;
}

// ---- 3c: target half of cat buffer: cat[b][n64][2048+c] = bf16(tgt) --------
__global__ void st_tgt_cat(const float* __restrict__ tgtp,
                           __bf16* __restrict__ cat) {
  long long id = (long long)blockIdx.x * blockDim.x + threadIdx.x;
  if (id >= (long long)BATCH * LP * CHN) return;
  int c = (int)(id % CHN);
  int n = (int)((id / CHN) % LP);
  int b = (int)(id / ((long long)CHN * LP));
  float val = (n < LL) ? tgtp[((size_t)(b * CHN + c)) * LL + n] : 0.f;
  cat[((size_t)(b * LP + n)) * CATK + CHN + c] = (__bf16)val;
}

// ---- 4: GEMM1 fold: flod_w(2048x18432) @ rr -> bf16 into cat[..][0:2048] ---
// Block = 64x64 output tile, 16 waves. TDM double-buffered 64x128 bf16 tiles:
//   buf0: A @0 (16KB), B @16384 ; buf1: A @32768, B @49152. (64KB dynamic LDS)
// 4 WMMA per wave between barriers; next slab's TDM overlaps current math.
__global__ __launch_bounds__(512) void st_gemm_fold(
    const __bf16* __restrict__ wbf, const __bf16* __restrict__ rr,
    const float* __restrict__ bias, __bf16* __restrict__ cat) {
  extern __shared__ char smem[];
  __bf16* sbase = (__bf16*)smem;
  int tid = threadIdx.x, lane = tid & 31, w = tid >> 5;
  int b = blockIdx.y, mblk = blockIdx.x;
  const __bf16* aT = wbf + (size_t)(mblk * 64) * CUK;
  const __bf16* bT = rr + ((size_t)(b * LP)) * CUK;
  if (w == 0) {   // prologue: slab ks=0 into buffer 0
    tdm_load_2d(0,     aT, KSTG, 64, KSTG, 64, CUK, 1);
    tdm_load_2d(16384, bT, KSTG, 64, KSTG, 64, CUK, 1);
    __builtin_amdgcn_s_wait_tensorcnt(0);
  }
  __syncthreads();
  int mt = w >> 2, nt = w & 3, lm = lane & 15, hi = lane >> 4;
  int aoff = (mt * 16 + lm) * KSTG + hi * 16;  // + {0,8} elements
  int boff = (nt * 16 + lm) * KSTG + hi * 16;  // + {0,8} elements
  v8f acc = {};
  for (int ks = 0; ks < CUK; ks += KSTG) {
    int cur = (ks / KSTG) & 1;
    if (w == 0 && ks + KSTG < CUK) {   // issue next slab (overlaps compute)
      int nb = cur ^ 1;
      tdm_load_2d(nb * 32768,         aT + ks + KSTG, KSTG, 64, KSTG, 64, CUK, 1);
      tdm_load_2d(nb * 32768 + 16384, bT + ks + KSTG, KSTG, 64, KSTG, 64, CUK, 1);
    }
    const __bf16* sA = sbase + cur * 16384;          // elements
    const __bf16* sB = sbase + cur * 16384 + 8192;
#pragma unroll
    for (int kk = 0; kk < KSTG; kk += 32) {
      v16bf av = frag_from_lds_A(sA + aoff + kk);
      v16bf bv = frag_from_lds_B(sB + boff + kk);
      acc = __builtin_amdgcn_wmma_f32_16x16x32_bf16(
          false, av, false, bv, (short)0, acc, false, false);
    }
    if (w == 0 && ks + KSTG < CUK) __builtin_amdgcn_s_wait_tensorcnt(0);
    __syncthreads();
  }
  int n  = (nt << 4) + lm;
  int mb = (mblk * 64) + (mt << 4) + (hi << 3);
  union { __bf16 h[8]; v4u u; } st;
  for (int r = 0; r < 8; ++r) st.h[r] = (__bf16)(acc[r] + bias[mb + r]);
  *(v4u*)(cat + ((size_t)(b * LP + n)) * CATK + mb) = st.u;  // transposed store
}

// ---- 5: GEMM2 conv + mask + residual -> f32 out (same TDM structure) -------
__global__ __launch_bounds__(512) void st_gemm_conv(
    const __bf16* __restrict__ cwbf, const __bf16* __restrict__ cat,
    const float* __restrict__ bias, const float* __restrict__ maxv,
    const float* __restrict__ tgtp, float* __restrict__ outp) {
  extern __shared__ char smem[];
  __bf16* sbase = (__bf16*)smem;
  int tid = threadIdx.x, lane = tid & 31, w = tid >> 5;
  int b = blockIdx.y, mblk = blockIdx.x;
  const __bf16* aT = cwbf + (size_t)(mblk * 64) * CATK;
  const __bf16* bT = cat + ((size_t)(b * LP)) * CATK;
  if (w == 0) {
    tdm_load_2d(0,     aT, KSTG, 64, KSTG, 64, CATK, 1);
    tdm_load_2d(16384, bT, KSTG, 64, KSTG, 64, CATK, 1);
    __builtin_amdgcn_s_wait_tensorcnt(0);
  }
  __syncthreads();
  int mt = w >> 2, nt = w & 3, lm = lane & 15, hi = lane >> 4;
  int aoff = (mt * 16 + lm) * KSTG + hi * 16;
  int boff = (nt * 16 + lm) * KSTG + hi * 16;
  v8f acc = {};
  for (int ks = 0; ks < CATK; ks += KSTG) {
    int cur = (ks / KSTG) & 1;
    if (w == 0 && ks + KSTG < CATK) {
      int nb = cur ^ 1;
      tdm_load_2d(nb * 32768,         aT + ks + KSTG, KSTG, 64, KSTG, 64, CATK, 1);
      tdm_load_2d(nb * 32768 + 16384, bT + ks + KSTG, KSTG, 64, KSTG, 64, CATK, 1);
    }
    const __bf16* sA = sbase + cur * 16384;
    const __bf16* sB = sbase + cur * 16384 + 8192;
#pragma unroll
    for (int kk = 0; kk < KSTG; kk += 32) {
      v16bf av = frag_from_lds_A(sA + aoff + kk);
      v16bf bv = frag_from_lds_B(sB + boff + kk);
      acc = __builtin_amdgcn_wmma_f32_16x16x32_bf16(
          false, av, false, bv, (short)0, acc, false, false);
    }
    if (w == 0 && ks + KSTG < CATK) __builtin_amdgcn_s_wait_tensorcnt(0);
    __syncthreads();
  }
  int n = (nt << 4) + lm;
  if (n < LL) {
    int mb = (mblk * 64) + (mt << 4) + (hi << 3);
    float msk = maxv[b * LL + n];
    for (int r = 0; r < 8; ++r) {
      int m = mb + r;
      size_t gi = ((size_t)(b * CHN + m)) * LL + n;
      outp[gi] = (acc[r] + bias[m]) * msk + tgtp[gi];
    }
  }
}

// ---------------------------------------------------------------------------
extern "C" void kernel_launch(void* const* d_in, const int* in_sizes, int n_in,
                              void* d_out, int out_size, void* d_ws, size_t ws_size,
                              hipStream_t stream) {
  const float* refp   = (const float*)d_in[0];
  const float* tgtp   = (const float*)d_in[1];
  const float* flod_w = (const float*)d_in[2];
  const float* flod_b = (const float*)d_in[3];
  const float* conv_w = (const float*)d_in[4];
  const float* conv_b = (const float*)d_in[5];
  float* outp = (float*)d_out;
  (void)in_sizes; (void)n_in; (void)out_size; (void)ws_size;

  char* ws = (char*)d_ws;
  size_t off = 0;
  auto carve = [&](size_t bytes) -> char* {
    char* p = ws + off;
    off = (off + bytes + 255) & ~(size_t)255;
    return p;
  };
  __bf16* wbf  = (__bf16*)carve((size_t)CHN * CUK * 2);        // 75.5 MB
  __bf16* cwbf = (__bf16*)carve((size_t)CHN * CATK * 2);       // 16.8 MB
  __bf16* rrbf = (__bf16*)carve((size_t)BATCH * LP * CUK * 2); // 151.0 MB
  __bf16* catb = (__bf16*)carve((size_t)BATCH * LP * CATK * 2);// 33.6 MB
  float*  s2   = (float*)carve((size_t)BATCH * 2 * LL * 4);
  float*  nrm  = (float*)carve((size_t)BATCH * 2 * LL * 4);
  float*  maxv = (float*)carve((size_t)BATCH * LL * 4);
  int*    maxi = (int*)carve((size_t)BATCH * LL * 4);

  // norms
  st_sumsq<<<dim3((BATCH * 2 * LL) / 8), 256, 0, stream>>>(refp, tgtp, s2);
  st_norms<<<dim3((BATCH * 2 * LL + 255) / 256), 256, 0, stream>>>(s2, nrm);
  // correlation + argmax (f32 WMMA, TDM staging, 48KB dynamic LDS)
  st_corr_argmax<<<dim3(BATCH), 512, 49152, stream>>>(refp, tgtp, nrm, maxv, maxi);
  // weight conversion (independent)
  st_cvt_bf16<<<dim3((unsigned)(((long long)CHN * CUK + 255) / 256)), 256, 0,
                stream>>>(flod_w, wbf, (long long)CHN * CUK);
  st_cvt_bf16<<<dim3((unsigned)(((long long)CHN * CATK + 255) / 256)), 256, 0,
                stream>>>(conv_w, cwbf, (long long)CHN * CATK);
  // gather rr + target half of cat
  st_gather_rr<<<dim3((unsigned)(((long long)BATCH * LP * CUK + 255) / 256)),
                 256, 0, stream>>>(refp, maxi, rrbf);
  st_tgt_cat<<<dim3((unsigned)(((long long)BATCH * LP * CHN + 255) / 256)),
               256, 0, stream>>>(tgtp, catb);
  // GEMM1: fold (bf16 WMMA, TDM double-buffered, 64KB dynamic LDS)
  st_gemm_fold<<<dim3(CHN / 64, BATCH), 512, 65536, stream>>>(wbf, rrbf,
                                                              flod_b, catb);
  // GEMM2: conv + mask + residual -> out
  st_gemm_conv<<<dim3(CHN / 64, BATCH), 512, 65536, stream>>>(cwbf, catb,
                                                              conv_b, maxv,
                                                              tgtp, outp);
}
